// ResBlockSparse_17437567221852
// MI455X (gfx1250) — matrix operand order repaired and checked by reading, not compiled, not run on hardware
//
#include <hip/hip_runtime.h>
#include <hip/hip_bf16.h>

typedef __attribute__((ext_vector_type(16))) __bf16 v16bf;
typedef __attribute__((ext_vector_type(8)))  float  v8f;

#define CCH 128            // channels
#define NK 27              // kernel offsets
#define TILE_M 128         // points per workgroup
#define LDS_STRIDE 136     // bf16 elems per LDS row (pad 8 to dodge bank conflicts)
#define STAT_BLOCKS 512

__device__ __forceinline__ unsigned short f32_to_bf16(float f) {
  unsigned int u = __float_as_uint(f);
  unsigned int r = u + 0x7FFFu + ((u >> 16) & 1u);   // round-to-nearest-even
  return (unsigned short)(r >> 16);
}

// ---------------- zero page (masked-gather source) ----------------
__global__ void zero_page(unsigned int* __restrict__ p) {
  p[threadIdx.x] = 0u;   // 64 threads x 4B = 256B
}

// ---------------- prep: feats f32 -> bf16 ----------------
__global__ __launch_bounds__(256) void cvt_feats(const float* __restrict__ x,
                                                 unsigned short* __restrict__ y,
                                                 int total) {
  int i = blockIdx.x * 256 + threadIdx.x;
  if (i < total) y[i] = f32_to_bf16(x[i]);
}

// ---------------- prep: W [27][128][128] f32 -> per-lane B fragments ----------
// wt index i = ((((k*4+kk)*8 + j)*32 + L)*16 + e)
// lane L: col = j*16 + (L&15); element e: ci = kk*32 + (L>>4)*16 + e
__global__ __launch_bounds__(256) void cvt_weights(const float* __restrict__ W,
                                                   unsigned short* __restrict__ wt) {
  int i = blockIdx.x * 256 + threadIdx.x;      // 27<<14 threads total
  int e  = i & 15;
  int L  = (i >> 4) & 31;
  int j  = (i >> 9) & 7;
  int kk = (i >> 12) & 3;
  int k  = i >> 14;
  if (k >= NK) return;
  int ci = kk * 32 + (L >> 4) * 16 + e;
  int co = j * 16 + (L & 15);
  wt[i] = f32_to_bf16(W[((size_t)k * CCH + ci) * CCH + co]);
}

// async global->LDS: 8 x b128 per thread (128B half-row), per-lane addresses
__device__ __forceinline__ void stage_async(const unsigned short* base,
                                            unsigned voff, unsigned laddr) {
  #pragma unroll
  for (int q = 0; q < 8; ++q) {
    asm volatile("global_load_async_to_lds_b128 %0, %1, %2"
                 :: "v"(laddr + (unsigned)q * 16u),
                    "v"(voff + (unsigned)q * 16u),
                    "s"(base)
                 : "memory");
  }
}

// ---------------- sparse conv: async-gather + WMMA-bf16, double-buffered -----
// xz points at the 256B zero page; real bf16 rows start at xz + 128 elems.
__global__ __launch_bounds__(256) void conv_wmma(const unsigned short* __restrict__ xz,
                                                 const unsigned short* __restrict__ wt,
                                                 const int* __restrict__ nbr,
                                                 float* __restrict__ out, int Npts) {
  __shared__ unsigned short lds[2][TILE_M * LDS_STRIDE];
  const int m0    = blockIdx.x * TILE_M;
  const int t     = threadIdx.x;
  const int lane  = t & 31;
  const int wave  = t >> 5;
  const int arow  = lane & 15;   // M row within 16-tile (N col for C/D)
  const int khalf = lane >> 4;   // K-half select for A/B, M-half for C/D

  v8f acc[8];
  #pragma unroll
  for (int j = 0; j < 8; ++j) { v8f z = {0.f,0.f,0.f,0.f,0.f,0.f,0.f,0.f}; acc[j] = z; }

  const int r    = t >> 1;       // staged row 0..127
  const int half = t & 1;

  auto issue_stage = [&](int k, int buf) {
    const int p = m0 + r;
    int idx = (p < Npts) ? nbr[(size_t)k * Npts + p] : -1;
    // masked rows read the zero page at region start (offset 0..255)
    unsigned voff = (idx >= 0 ? ((unsigned)idx * 256u + 256u) : 0u)
                  + (unsigned)half * 128u;
    unsigned laddr = (unsigned)(unsigned long long)(const void*)
                     &lds[buf][r * LDS_STRIDE + half * 64];
    stage_async(xz, voff, laddr);
  };

  issue_stage(0, 0);

  for (int k = 0; k < NK; ++k) {
    asm volatile("s_wait_asynccnt 0x0" ::: "memory");   // my stage(k) landed
    __syncthreads();                                    // everyone's stage(k) + compute(k-1) done
    if (k + 1 < NK) issue_stage(k + 1, (k + 1) & 1);    // overlaps with compute(k)

    const unsigned short* cur = lds[k & 1];

    #pragma unroll
    for (int kk = 0; kk < 4; ++kk) {
      // A fragment (16x32 bf16) from LDS: per-lane K-halves, two b128 loads
      union { uint4 u[2]; v16bf v; } ua;
      const unsigned short* ap =
          &cur[(wave * 16 + arow) * LDS_STRIDE + kk * 32 + khalf * 16];
      ua.u[0] = *(const uint4*)ap;
      ua.u[1] = *(const uint4*)(ap + 8);

      #pragma unroll
      for (int j = 0; j < 8; ++j) {
        // B fragment (32x16 bf16) pre-packed per lane: 32 contiguous bytes
        union { uint4 u[2]; v16bf v; } ub;
        const unsigned short* bp =
            wt + (((size_t)((k * 4 + kk) * 8 + j) * 32 + lane) * 16);
        ub.u[0] = *(const uint4*)bp;
        ub.u[1] = *(const uint4*)(bp + 8);
        acc[j] = __builtin_amdgcn_wmma_f32_16x16x32_bf16(
            false, ua.v, false, ub.v, (short)0, acc[j], false, false);
      }
    }
  }

  // C/D layout: VGPR r -> M = khalf*8 + r, N = arow
  #pragma unroll
  for (int j = 0; j < 8; ++j) {
    #pragma unroll
    for (int rr = 0; rr < 8; ++rr) {
      int m = m0 + wave * 16 + khalf * 8 + rr;
      if (m < Npts) out[(size_t)m * CCH + j * 16 + arow] = acc[j][rr];
    }
  }
}

// ---------------- GroupNorm stage 1: deterministic per-block partials --------
__global__ __launch_bounds__(256) void gn_stats(const float* __restrict__ x,
                                                float* __restrict__ part, int Npts) {
  __shared__ float rs[256], rss[256];
  const int t = threadIdx.x;
  const int c = t & 127;
  float s = 0.f, ss = 0.f;
  for (int rr = blockIdx.x * 2 + (t >> 7); rr < Npts; rr += gridDim.x * 2) {
    float v = x[(size_t)rr * CCH + c];
    s += v; ss += v * v;
  }
  rs[t] = s; rss[t] = ss;
  __syncthreads();
  if (t < 128) { rs[t] += rs[t + 128]; rss[t] += rss[t + 128]; }
  __syncthreads();
  if (t < 8) {
    float a = 0.f, b = 0.f;
    for (int q = 0; q < 16; ++q) { a += rs[t * 16 + q]; b += rss[t * 16 + q]; }
    part[blockIdx.x * 16 + t]     = a;
    part[blockIdx.x * 16 + 8 + t] = b;
  }
}

// ---------------- GroupNorm stage 2: fixed-order finalize -> (mu, rsqrt) ----
__global__ void gn_final(const float* __restrict__ part, int nblk,
                         float* __restrict__ stats, float invcount) {
  __shared__ float sh[16];
  int t = threadIdx.x;  // 16 threads
  float a = 0.f;
  for (int b = 0; b < nblk; ++b) a += part[b * 16 + t];
  sh[t] = a;
  __syncthreads();
  if (t < 8) {
    float mu  = sh[t] * invcount;
    float var = sh[8 + t] * invcount - mu * mu;
    stats[t * 2]     = mu;
    stats[t * 2 + 1] = rsqrtf(var + 1e-5f);
  }
}

// ---------------- normalize + affine + ReLU -> bf16 (conv1 input) -----------
__global__ __launch_bounds__(256) void gn_apply_bf16(const float* __restrict__ x,
                                                     const float* __restrict__ stats,
                                                     const float* __restrict__ gamma,
                                                     const float* __restrict__ beta,
                                                     unsigned short* __restrict__ y,
                                                     int total) {
  int i = blockIdx.x * 256 + threadIdx.x;
  if (i >= total) return;
  int c = i & 127, g = c >> 4;
  float mu = stats[g * 2], inv = stats[g * 2 + 1];
  float v = (x[i] - mu) * inv * gamma[c] + beta[c];
  y[i] = f32_to_bf16(v > 0.f ? v : 0.f);
}

// ---------------- normalize + affine + ReLU + residual -> f32 out -----------
__global__ __launch_bounds__(256) void gn_apply_res(const float* __restrict__ x,
                                                    const float* __restrict__ stats,
                                                    const float* __restrict__ gamma,
                                                    const float* __restrict__ beta,
                                                    const float* __restrict__ feats,
                                                    float* __restrict__ out, int total) {
  int i = blockIdx.x * 256 + threadIdx.x;
  if (i >= total) return;
  int c = i & 127, g = c >> 4;
  float mu = stats[g * 2], inv = stats[g * 2 + 1];
  float v = (x[i] - mu) * inv * gamma[c] + beta[c];
  out[i] = feats[i] + (v > 0.f ? v : 0.f);
}

extern "C" void kernel_launch(void* const* d_in, const int* in_sizes, int n_in,
                              void* d_out, int out_size, void* d_ws, size_t ws_size,
                              hipStream_t stream) {
  const float* feats = (const float*)d_in[0];
  const int*   nbr   = (const int*)d_in[1];
  const float* W0    = (const float*)d_in[2];
  const float* g0    = (const float*)d_in[3];
  const float* b0    = (const float*)d_in[4];
  const float* W1    = (const float*)d_in[5];
  const float* g1    = (const float*)d_in[6];
  const float* b1    = (const float*)d_in[7];
  float* out = (float*)d_out;

  const int Npts  = in_sizes[0] / CCH;
  const int total = Npts * CCH;

  // workspace carve-up (256B aligned)
  char* ws = (char*)d_ws;
  size_t off = 0;
  unsigned short* xz  = (unsigned short*)(ws + off);        // [256B zero page][bf16 acts]
  unsigned short* xbf = xz + 128;                           // data region (reused for h)
  off += (256 + (size_t)total * 2 + 255) & ~255ull;
  float* accbuf       = (float*)(ws + off);          off += ((size_t)total * 4 + 255) & ~255ull;
  const size_t wtElems = (size_t)NK * 4 * 8 * 32 * 16;
  unsigned short* wt0 = (unsigned short*)(ws + off); off += (wtElems * 2 + 255) & ~255ull;
  unsigned short* wt1 = (unsigned short*)(ws + off); off += (wtElems * 2 + 255) & ~255ull;
  float* part         = (float*)(ws + off);          off += ((size_t)STAT_BLOCKS * 16 * 4 + 255) & ~255ull;
  float* stats0       = (float*)(ws + off);          off += 256;
  float* stats1       = (float*)(ws + off);          off += 256;

  const int ewBlocks   = (total + 255) / 256;
  const int wBlocks    = (int)((wtElems + 255) / 256);
  const int convBlocks = (Npts + TILE_M - 1) / TILE_M;
  const float invcount = 1.0f / ((float)Npts * 16.0f);

  // prep
  zero_page<<<1, 64, 0, stream>>>((unsigned int*)xz);
  cvt_feats<<<ewBlocks, 256, 0, stream>>>(feats, xbf, total);
  cvt_weights<<<wBlocks, 256, 0, stream>>>(W0, wt0);
  cvt_weights<<<wBlocks, 256, 0, stream>>>(W1, wt1);

  // conv0 -> GN0 -> ReLU -> h (bf16, written back over xbf)
  conv_wmma<<<convBlocks, 256, 0, stream>>>(xz, wt0, nbr, accbuf, Npts);
  gn_stats<<<STAT_BLOCKS, 256, 0, stream>>>(accbuf, part, Npts);
  gn_final<<<1, 16, 0, stream>>>(part, STAT_BLOCKS, stats0, invcount);
  gn_apply_bf16<<<ewBlocks, 256, 0, stream>>>(accbuf, stats0, g0, b0, xbf, total);

  // conv1 -> GN1 -> ReLU -> residual add
  conv_wmma<<<convBlocks, 256, 0, stream>>>(xz, wt1, nbr, accbuf, Npts);
  gn_stats<<<STAT_BLOCKS, 256, 0, stream>>>(accbuf, part, Npts);
  gn_final<<<1, 16, 0, stream>>>(part, STAT_BLOCKS, stats1, invcount);
  gn_apply_res<<<ewBlocks, 256, 0, stream>>>(accbuf, stats1, g1, b1, feats, out, total);
}
